// FSAPhonotacticsModel_88270167867764
// MI455X (gfx1250) — matrix operand order repaired and checked by reading, not compiled, not run on hardware
//
#include <hip/hip_runtime.h>
#include <hip/hip_bf16.h>

typedef __attribute__((ext_vector_type(16))) _Float16 v16h;
typedef __attribute__((ext_vector_type(8)))  _Float16 v8h;
typedef __attribute__((ext_vector_type(8)))  float    v8f;

#define QN     128
#define SN     96
#define TN     48
#define BATCH  2048
#define SHIFT  12.0f                 // P stored as exp(A_norm) * 2^12 (exact, subtracted at the end)
#define LOG2E  1.4426950408889634f
#define LN2    0.6931471805599453f

// ---------------------------------------------------------------------------
// Kernel 1: per-source-state logsumexp (fp32) and build the transposed,
// scaled f16 transition tensor  PT[s][q'][q] = exp(A[q][s][q'] - logZ[q]) * 2^12
// One workgroup per q (128 WGs). A slice per q = 96*128 floats (48 KB).
// ---------------------------------------------------------------------------
__global__ void fsa_prep(const float* __restrict__ A, _Float16* __restrict__ PT) {
    __shared__ float red[256];
    const int q   = blockIdx.x;
    const int tid = threadIdx.x;
    const float* Aq = A + (size_t)q * (SN * QN);

    // pass 1: max
    float m = -1e30f;
    for (int i = tid; i < SN * QN; i += 256) m = fmaxf(m, Aq[i]);
    red[tid] = m; __syncthreads();
    for (int off = 128; off > 0; off >>= 1) {
        if (tid < off) red[tid] = fmaxf(red[tid], red[tid + off]);
        __syncthreads();
    }
    m = red[0]; __syncthreads();

    // pass 2: sum of exp
    float s = 0.f;
    for (int i = tid; i < SN * QN; i += 256) s += __expf(Aq[i] - m);
    red[tid] = s; __syncthreads();
    for (int off = 128; off > 0; off >>= 1) {
        if (tid < off) red[tid] += red[tid + off];
        __syncthreads();
    }
    const float logZ = m + __logf(red[0]);

    // pass 3: scaled exp, transposed scatter into PT (one-time ~3 MB, stays in L2)
    for (int i = tid; i < SN * QN; i += 256) {
        float v = exp2f((Aq[i] - logZ) * LOG2E + SHIFT);
        int sym = i >> 7;            // s
        int qp  = i & 127;           // q'
        PT[((size_t)sym * QN + qp) * QN + q] = (_Float16)v;
    }
}

// ---------------------------------------------------------------------------
// Kernel 2: scaled forward algorithm. One wave32 per sequence, 8 waves/block.
// Per step: la = alpha(1x128) @ P_tok(128x128) via 8x4 = 32 WMMA f16 tiles.
// Software-pipelined: the 8-tile B load group for K-block kt+1 (or for the
// NEXT token's kt=0 group) is issued before the WMMAs of kt — the B stream
// never depends on alpha, so L2 latency is hidden with registers, which is
// the right trade at 2048 total waves (occupancy-starved on MI455X).
// ---------------------------------------------------------------------------
__global__ void __launch_bounds__(256) fsa_forward(const _Float16* __restrict__ PT,
                                                   const int* __restrict__ xs,
                                                   float* __restrict__ out) {
    __shared__ _Float16 sAlpha[8][QN];
    const int lane = threadIdx.x & 31;
    const int wv   = threadIdx.x >> 5;
    const int b    = blockIdx.x * 8 + wv;
    _Float16* al = sAlpha[wv];

    // alpha_0 = one-hot(state 0)
    for (int i = lane; i < QN; i += 32) al[i] = (_Float16)(i == 0 ? 1.0f : 0.0f);

    // token sequence is wave-uniform: park it in 2 VGPRs, broadcast via readlane
    const int* xb = xs + b * TN;
    int tv0 = xb[lane];
    int tv1 = (lane < (TN - 32)) ? xb[32 + lane] : 0;

    const int nof  = lane & 15;          // N index within a 16-wide tile
    const int koff = (lane >> 4) * 16;   // B-operand: K sub-block (0 or 16)
    const int aoff = (lane >> 4) * 16;   // A-operand: K sub-block (0 or 16)

    // B-tile group loader: 8 N-tiles of the given 32-wide K block, one v16h
    // (2x global_load_b128, contiguous) per lane per tile.
    auto ldB = [&](const _Float16* Ps, int kt, v16h* bop) {
#pragma unroll
        for (int nt = 0; nt < 8; ++nt)
            bop[nt] = *(const v16h*)(Ps + (size_t)(nt * 16 + nof) * QN + koff + kt * 32);
    };

    int tok = __builtin_amdgcn_readlane(tv0, 0);
    const _Float16* Ps = PT + (size_t)tok * (QN * QN);

    v16h b0[8], b1[8];
    ldB(Ps, 0, b0);                      // prime the pipeline: t=0, kt=0

    float L = 0.0f;

    for (int t = 0; t < TN; ++t) {
        // next token (clamped at the end; duplicate load group is harmless)
        const int tn = (t + 1 < TN) ? (t + 1) : t;
        const int tokn = (tn < 32) ? __builtin_amdgcn_readlane(tv0, tn)
                                   : __builtin_amdgcn_readlane(tv1, tn - 32);
        const _Float16* Psn = PT + (size_t)tokn * (QN * QN);

        // A operands (alpha) from LDS, in contiguous K order (as B): lane<16 holds
        // K{0..15}, lane>=16 holds K{16..31}. All M rows = alpha.
        v16h aop[4];
#pragma unroll
        for (int kt = 0; kt < 4; ++kt) {
            v8h lo = *(const v8h*)(al + kt * 32 + aoff);
            v8h hi = *(const v8h*)(al + kt * 32 + aoff + 8);
            aop[kt] = __builtin_shufflevector(lo, hi,
                        0,1,2,3,4,5,6,7,8,9,10,11,12,13,14,15);
        }

        v8f acc[8] = {};

        // kt = 0 : consume b0, prefetch kt=1 into b1
        ldB(Ps, 1, b1);
#pragma unroll
        for (int nt = 0; nt < 8; ++nt)
            acc[nt] = __builtin_amdgcn_wmma_f32_16x16x32_f16(
                          false, aop[0], false, b0[nt], (short)0, acc[nt], false, false);
        // kt = 1 : consume b1, prefetch kt=2 into b0
        ldB(Ps, 2, b0);
#pragma unroll
        for (int nt = 0; nt < 8; ++nt)
            acc[nt] = __builtin_amdgcn_wmma_f32_16x16x32_f16(
                          false, aop[1], false, b1[nt], (short)0, acc[nt], false, false);
        // kt = 2 : consume b0, prefetch kt=3 into b1
        ldB(Ps, 3, b1);
#pragma unroll
        for (int nt = 0; nt < 8; ++nt)
            acc[nt] = __builtin_amdgcn_wmma_f32_16x16x32_f16(
                          false, aop[2], false, b0[nt], (short)0, acc[nt], false, false);
        // kt = 3 : consume b1, prefetch NEXT step's kt=0 into b0
        ldB(Psn, 0, b0);
#pragma unroll
        for (int nt = 0; nt < 8; ++nt)
            acc[nt] = __builtin_amdgcn_wmma_f32_16x16x32_f16(
                          false, aop[3], false, b1[nt], (short)0, acc[nt], false, false);

        // c_t: D row M=0 -> lane n holds la[nt*16 + (n&15)]; each 16-lane half
        // holds the complete la vector, so reduce inside halves.
        float la[8];
        float csum = 0.f;
#pragma unroll
        for (int nt = 0; nt < 8; ++nt) { la[nt] = acc[nt][0]; csum += la[nt]; }
        csum += __shfl_xor(csum, 1);
        csum += __shfl_xor(csum, 2);
        csum += __shfl_xor(csum, 4);
        csum += __shfl_xor(csum, 8);

        L += __logf(csum);
        // v_rcp_f32: 1-ulp scale error is absorbed exactly by accumulating
        // log of the actual sums.
        float r = __builtin_amdgcn_rcpf(csum);
        if (lane < 16) {
#pragma unroll
            for (int nt = 0; nt < 8; ++nt)
                al[nt * 16 + lane] = (_Float16)(la[nt] * r);
        }
        // alpha store -> next-iteration A-operand load (same wave): wait LDS
        asm volatile("s_wait_dscnt 0" ::: "memory");

        Ps = Psn;
    }

    if (lane == 0) out[b] = L - (float)TN * SHIFT * LN2;
}

// ---------------------------------------------------------------------------
// d_in[0]: A   [128*96*128] float32
// d_in[1]: xs  [2048*48]    int32
// d_out  : [2048] float32
// d_ws   : >= 96*128*128*2 = 3,145,728 bytes (f16 transition tensor, L2-resident)
// ---------------------------------------------------------------------------
extern "C" void kernel_launch(void* const* d_in, const int* in_sizes, int n_in,
                              void* d_out, int out_size, void* d_ws, size_t ws_size,
                              hipStream_t stream) {
    const float* A  = (const float*)d_in[0];
    const int*   xs = (const int*)d_in[1];
    float*       o  = (float*)d_out;
    _Float16*    PT = (_Float16*)d_ws;

    fsa_prep<<<QN, 256, 0, stream>>>(A, PT);
    fsa_forward<<<BATCH / 8, 256, 0, stream>>>(PT, xs, o);
}
